// PairedMultiheadAttention_4604204941923
// MI455X (gfx1250) — matrix operand order repaired and checked by reading, not compiled, not run on hardware
//
#include <hip/hip_runtime.h>

// Problem constants
#define LQ   2048
#define NB   4
#define EE   768
#define HH   12
#define DDIM 64
#define E3   2304
#define LN   8192      // L*N rows of the (L,N,E) activations
#define DCAT 128       // concatenated head dim (x||y)
#define NHH  48        // N*H
#define PSTR 40        // padded LDS row stride for P tiles (80B = 5x16B: aligned + conflict-free)
#define QSCALE 0.08838834764831845f  // 1/sqrt(D * SCALE_FACTOR) = 1/sqrt(128)

typedef __bf16 bf16_t;
typedef bf16_t bf16x8  __attribute__((ext_vector_type(8)));
typedef bf16_t bf16x16 __attribute__((ext_vector_type(16)));
typedef float  f32x8   __attribute__((ext_vector_type(8)));

__device__ __forceinline__ bf16_t f2bf(float f) {
  unsigned u = __builtin_bit_cast(unsigned, f);
  unsigned r = u + 0x7FFFu + ((u >> 16) & 1u);          // round-to-nearest-even
  return __builtin_bit_cast(bf16_t, (unsigned short)(r >> 16));
}

__device__ __forceinline__ bf16x16 cat8(bf16x8 lo, bf16x8 hi) {
  return __builtin_shufflevector(lo, hi, 0,1,2,3,4,5,6,7,8,9,10,11,12,13,14,15);
}

// A-fragment: 16x32 bf16 tile from row-major storage (row stride ld elements).
// ISA layout: lanes 0-15 rows 0-15 hold K [0,8)+[16,24); lanes 16-31 hold K [8,16)+[24,32).
__device__ __forceinline__ bf16x16 load_a_frag(const bf16_t* __restrict__ base, int ld) {
  int lane = threadIdx.x & 31;
  const bf16_t* p = base + (lane & 15) * (size_t)ld + ((lane >> 4) << 3);
  return cat8(*(const bf16x8*)p, *(const bf16x8*)(p + 16));
}

// B-fragment: 32x16 bf16 tile; storage is "N x K row-major" (column n of B contiguous in K).
// ISA layout: lanes 0-15 hold K [0,8)+[16,24) of column (lane&15); lanes 16-31 hold K [8,16)+[24,32).
__device__ __forceinline__ bf16x16 load_b_frag(const bf16_t* __restrict__ base, int ld) {
  int lane = threadIdx.x & 31;
  const bf16_t* p = base + (lane & 15) * (size_t)ld + ((lane >> 4) << 3);
  return cat8(*(const bf16x8*)p, *(const bf16x8*)(p + 16));
}

__device__ __forceinline__ f32x8 wmma_bf16(bf16x16 a, bf16x16 b, f32x8 c) {
  return __builtin_amdgcn_wmma_f32_16x16x32_bf16(false, a, false, b, (short)0, c, false, false);
}

// ---------------------------------------------------------------- fp32 -> bf16
__global__ void __launch_bounds__(256, 1)
cvt_bf16_kernel(const float* __restrict__ src, bf16_t* __restrict__ dst, int n) {
  int i = blockIdx.x * blockDim.x + threadIdx.x;
  if (i < n) dst[i] = f2bf(src[i]);
}

// ------------------------------------------------------- fused QKV projection
// 32x64 output tile per wave: 2 A-frags x 4 B-frags -> 8 WMMAs per k-step.
// q -> qcat (N,H,L,128) scaled; k -> kcat (N,H,L,128); v -> vT (N,H,D,L).
__global__ void __launch_bounds__(32, 1)
qkv_kernel(const bf16_t* __restrict__ ab, const bf16_t* __restrict__ wb,
           const float* __restrict__ bias,
           bf16_t* __restrict__ qcat, bf16_t* __restrict__ kcat,
           bf16_t* __restrict__ vT, int soff) {
  const int c0 = blockIdx.x * 64;       // 64-aligned => single segment & single head per block
  const int m0 = blockIdx.y * 32;
  const int lane = threadIdx.x;

  f32x8 acc[2][4];
  #pragma unroll
  for (int mt = 0; mt < 2; ++mt)
    #pragma unroll
    for (int t = 0; t < 4; ++t)
      #pragma unroll
      for (int i = 0; i < 8; ++i) acc[mt][t][i] = 0.f;

  #pragma unroll 2
  for (int k0 = 0; k0 < EE; k0 += 32) {
    bf16x16 a0 = load_a_frag(ab + (size_t)m0 * EE + k0, EE);
    bf16x16 a1 = load_a_frag(ab + (size_t)(m0 + 16) * EE + k0, EE);
    #pragma unroll
    for (int t = 0; t < 4; ++t) {
      bf16x16 b = load_b_frag(wb + (size_t)(c0 + 16 * t) * EE + k0, EE);
      acc[0][t] = wmma_bf16(a0, b, acc[0][t]);
      acc[1][t] = wmma_bf16(a1, b, acc[1][t]);
    }
  }

  const int col = lane & 15;
  const int rhi = (lane >> 4) << 3;
  const int seg = (c0 >= 2 * EE) ? 2 : (c0 >= EE ? 1 : 0);
  const int cc0 = c0 - seg * EE;        // multiple of 64
  const int h = cc0 >> 6;

  #pragma unroll
  for (int mt = 0; mt < 2; ++mt) {
    #pragma unroll
    for (int t = 0; t < 4; ++t) {
      const int d = t * 16 + col;       // 0..63 within head
      const float bv = bias[c0 + t * 16 + col];
      #pragma unroll
      for (int i = 0; i < 8; ++i) {
        int m = m0 + mt * 16 + rhi + i;
        int l = m >> 2, n = m & 3;
        float v = acc[mt][t][i] + bv;
        size_t nh = (size_t)(n * HH + h);
        if (seg == 0) {
          qcat[(nh * LQ + l) * DCAT + soff + d] = f2bf(v * QSCALE);
        } else if (seg == 1) {
          kcat[(nh * LQ + l) * DCAT + soff + d] = f2bf(v);
        } else {
          vT[(nh * DDIM + d) * (size_t)LQ + l] = f2bf(v);
        }
      }
    }
  }
}

// ------------------------------------------- flash attention over both streams
// Block = 4 waves, 128 consecutive queries of one (n,h); per-step __syncthreads
// keeps waves in lockstep so their identical K/V loads hit the WGP cache.
// Each wave: M=32 (2 query tiles sharing every K/V fragment), 32 keys per step.
__global__ void __launch_bounds__(128, 1)
attn_kernel(const bf16_t* __restrict__ qcat, const bf16_t* __restrict__ kcat,
            const bf16_t* __restrict__ vtx, const bf16_t* __restrict__ vty,
            bf16_t* __restrict__ oxb, bf16_t* __restrict__ oyb) {
  __shared__ __align__(16) bf16_t Pl[4 * 2 * 16 * PSTR];   // [wave][qtile][16 x PSTR]
  const int lane = threadIdx.x & 31;
  const int wv   = threadIdx.x >> 5;
  const int nh = blockIdx.y;               // 0..47
  const int n = nh / HH, h = nh % HH;
  const int l0 = blockIdx.x * 128 + wv * 32;

  const bf16_t* qb  = qcat + (size_t)nh * LQ * DCAT;
  const bf16_t* kbp = kcat + (size_t)nh * LQ * DCAT;
  const bf16_t* vxb = vtx + (size_t)nh * DDIM * LQ;
  const bf16_t* vyb = vty + (size_t)nh * DDIM * LQ;
  bf16_t* Pw = &Pl[wv * (2 * 16 * PSTR)];

  bf16x16 qf[2][4];
  #pragma unroll
  for (int qt = 0; qt < 2; ++qt)
    #pragma unroll
    for (int j = 0; j < 4; ++j)
      qf[qt][j] = load_a_frag(qb + (size_t)(l0 + qt * 16) * DCAT + j * 32, DCAT);

  f32x8 ax[2][4], ay[2][4];
  float mrow[2][8], lrow[2][8];
  #pragma unroll
  for (int qt = 0; qt < 2; ++qt) {
    #pragma unroll
    for (int t = 0; t < 4; ++t)
      #pragma unroll
      for (int i = 0; i < 8; ++i) { ax[qt][t][i] = 0.f; ay[qt][t][i] = 0.f; }
    #pragma unroll
    for (int i = 0; i < 8; ++i) { mrow[qt][i] = -3.0e38f; lrow[qt][i] = 0.f; }
  }

  const int col = lane & 15;
  const int rhi = (lane >> 4) << 3;

  for (int s0 = 0; s0 < LQ; s0 += 32) {
    // ---- prefetch next step's K/V tiles (global_prefetch_b8; per-lane lines) ----
    if (s0 + 32 < LQ) {
      const bf16_t* kn = kbp + (size_t)(s0 + 32) * DCAT + (size_t)lane * DCAT;
      __builtin_prefetch(kn, 0, 1);
      __builtin_prefetch(kn + 64, 0, 1);
      const bf16_t* vxn = vxb + (size_t)lane * LQ + s0 + 32;
      __builtin_prefetch(vxn, 0, 1);
      __builtin_prefetch(vxn + 32 * (size_t)LQ, 0, 1);
      const bf16_t* vyn = vyb + (size_t)lane * LQ + s0 + 32;
      __builtin_prefetch(vyn, 0, 1);
      __builtin_prefetch(vyn + 32 * (size_t)LQ, 0, 1);
    }

    // ---- scores: 2 qtiles x 2 stiles over concatenated D=128 (16 WMMAs) ----
    f32x8 sc[2][2];
    #pragma unroll
    for (int qt = 0; qt < 2; ++qt)
      #pragma unroll
      for (int st = 0; st < 2; ++st)
        #pragma unroll
        for (int i = 0; i < 8; ++i) sc[qt][st][i] = 0.f;
    #pragma unroll
    for (int j = 0; j < 4; ++j) {
      bf16x16 b0 = load_b_frag(kbp + (size_t)s0 * DCAT + j * 32, DCAT);
      bf16x16 b1 = load_b_frag(kbp + (size_t)(s0 + 16) * DCAT + j * 32, DCAT);
      #pragma unroll
      for (int qt = 0; qt < 2; ++qt) {
        sc[qt][0] = wmma_bf16(qf[qt][j], b0, sc[qt][0]);
        sc[qt][1] = wmma_bf16(qf[qt][j], b1, sc[qt][1]);
      }
    }

    // ---- online softmax per qtile (rows live across 16 lanes) ----
    #pragma unroll
    for (int qt = 0; qt < 2; ++qt) {
      float scl[8];
      #pragma unroll
      for (int i = 0; i < 8; ++i) {
        float t = fmaxf(sc[qt][0][i], sc[qt][1][i]);
        #pragma unroll
        for (int off = 1; off < 16; off <<= 1) t = fmaxf(t, __shfl_xor(t, off, 16));
        float mn = fmaxf(mrow[qt][i], t);
        scl[i] = __expf(mrow[qt][i] - mn);
        mrow[qt][i] = mn;
      }
      #pragma unroll
      for (int i = 0; i < 8; ++i) {
        sc[qt][0][i] = __expf(sc[qt][0][i] - mrow[qt][i]);
        sc[qt][1][i] = __expf(sc[qt][1][i] - mrow[qt][i]);
        float t = sc[qt][0][i] + sc[qt][1][i];
        #pragma unroll
        for (int off = 1; off < 16; off <<= 1) t += __shfl_xor(t, off, 16);
        lrow[qt][i] = lrow[qt][i] * scl[i] + t;
      }
      #pragma unroll
      for (int t = 0; t < 4; ++t)
        #pragma unroll
        for (int i = 0; i < 8; ++i) { ax[qt][t][i] *= scl[i]; ay[qt][t][i] *= scl[i]; }
    }

    // ---- P: C-layout -> A-layout via padded LDS ----
    __syncthreads();
    #pragma unroll
    for (int qt = 0; qt < 2; ++qt) {
      bf16_t* Pq = Pw + qt * (16 * PSTR);
      #pragma unroll
      for (int i = 0; i < 8; ++i) {
        Pq[(rhi + i) * PSTR + col]      = f2bf(sc[qt][0][i]);
        Pq[(rhi + i) * PSTR + col + 16] = f2bf(sc[qt][1][i]);
      }
    }
    __syncthreads();

    bf16x16 pf[2];
    {
      int r = lane & 15, kofs = (lane >> 4) << 3;
      #pragma unroll
      for (int qt = 0; qt < 2; ++qt) {
        const bf16_t* Pq = Pw + qt * (16 * PSTR);
        bf16x8 plo = *(const bf16x8*)&Pq[r * PSTR + kofs];
        bf16x8 phi = *(const bf16x8*)&Pq[r * PSTR + kofs + 16];
        pf[qt] = cat8(plo, phi);
      }
    }

    // ---- P(16x32) x Vt(32x16): both streams, 4 d-tiles, 2 qtiles (16 WMMAs) ----
    #pragma unroll
    for (int dt = 0; dt < 4; ++dt) {
      bf16x16 vxf = load_b_frag(vxb + (size_t)(dt * 16) * LQ + s0, LQ);
      ax[0][dt] = wmma_bf16(pf[0], vxf, ax[0][dt]);
      ax[1][dt] = wmma_bf16(pf[1], vxf, ax[1][dt]);
      bf16x16 vyf = load_b_frag(vyb + (size_t)(dt * 16) * LQ + s0, LQ);
      ay[0][dt] = wmma_bf16(pf[0], vyf, ay[0][dt]);
      ay[1][dt] = wmma_bf16(pf[1], vyf, ay[1][dt]);
    }
  }

  // ---- normalize and store to (L,N,E) bf16 staging buffers ----
  #pragma unroll
  for (int qt = 0; qt < 2; ++qt) {
    #pragma unroll
    for (int i = 0; i < 8; ++i) {
      float inv = 1.0f / lrow[qt][i];
      int l = l0 + qt * 16 + rhi + i;
      size_t rowbase = ((size_t)l * NB + n) * EE + h * DDIM;
      #pragma unroll
      for (int t = 0; t < 4; ++t) {
        oxb[rowbase + t * 16 + col] = f2bf(ax[qt][t][i] * inv);
        oyb[rowbase + t * 16 + col] = f2bf(ay[qt][t][i] * inv);
      }
    }
  }
}

// ------------------------------------------------------------ out projection
// 32x64 output tile per wave, same blocking as qkv_kernel.
__global__ void __launch_bounds__(32, 1)
outproj_kernel(const bf16_t* __restrict__ ob, const bf16_t* __restrict__ wb,
               const float* __restrict__ bias, float* __restrict__ out) {
  const int c0 = blockIdx.x * 64;
  const int m0 = blockIdx.y * 32;
  const int lane = threadIdx.x;

  f32x8 acc[2][4];
  #pragma unroll
  for (int mt = 0; mt < 2; ++mt)
    #pragma unroll
    for (int t = 0; t < 4; ++t)
      #pragma unroll
      for (int i = 0; i < 8; ++i) acc[mt][t][i] = 0.f;

  #pragma unroll 2
  for (int k0 = 0; k0 < EE; k0 += 32) {
    bf16x16 a0 = load_a_frag(ob + (size_t)m0 * EE + k0, EE);
    bf16x16 a1 = load_a_frag(ob + (size_t)(m0 + 16) * EE + k0, EE);
    #pragma unroll
    for (int t = 0; t < 4; ++t) {
      bf16x16 b = load_b_frag(wb + (size_t)(c0 + 16 * t) * EE + k0, EE);
      acc[0][t] = wmma_bf16(a0, b, acc[0][t]);
      acc[1][t] = wmma_bf16(a1, b, acc[1][t]);
    }
  }

  const int col = lane & 15;
  const int rhi = (lane >> 4) << 3;
  #pragma unroll
  for (int mt = 0; mt < 2; ++mt) {
    #pragma unroll
    for (int t = 0; t < 4; ++t) {
      const float bv = bias[c0 + t * 16 + col];
      #pragma unroll
      for (int i = 0; i < 8; ++i)
        out[(size_t)(m0 + mt * 16 + rhi + i) * EE + c0 + t * 16 + col] = acc[mt][t][i] + bv;
    }
  }
}

// ---------------------------------------------------------------------------
extern "C" void kernel_launch(void* const* d_in, const int* in_sizes, int n_in,
                              void* d_out, int out_size, void* d_ws, size_t ws_size,
                              hipStream_t stream) {
  const float* x       = (const float*)d_in[0];
  const float* y       = (const float*)d_in[1];
  const float* in_w_x  = (const float*)d_in[2];
  const float* in_b_x  = (const float*)d_in[3];
  const float* in_w_y  = (const float*)d_in[4];
  const float* in_b_y  = (const float*)d_in[5];
  const float* out_w_x = (const float*)d_in[6];
  const float* out_b_x = (const float*)d_in[7];
  const float* out_w_y = (const float*)d_in[8];
  const float* out_b_y = (const float*)d_in[9];
  float* out = (float*)d_out;

  char* ws = (char*)d_ws;
  size_t off = 0;
  auto alloc_bf = [&](size_t nelem) -> bf16_t* {
    bf16_t* p = (bf16_t*)(ws + off);
    off += (nelem * sizeof(bf16_t) + 255) & ~(size_t)255;
    return p;
  };
  bf16_t* xb   = alloc_bf((size_t)LN * EE);
  bf16_t* yb   = alloc_bf((size_t)LN * EE);
  bf16_t* wixb = alloc_bf((size_t)E3 * EE);
  bf16_t* wiyb = alloc_bf((size_t)E3 * EE);
  bf16_t* woxb = alloc_bf((size_t)EE * EE);
  bf16_t* woyb = alloc_bf((size_t)EE * EE);
  bf16_t* qcat = alloc_bf((size_t)NHH * LQ * DCAT);
  bf16_t* kcat = alloc_bf((size_t)NHH * LQ * DCAT);
  bf16_t* vtx  = alloc_bf((size_t)NHH * DDIM * LQ);
  bf16_t* vty  = alloc_bf((size_t)NHH * DDIM * LQ);
  bf16_t* oxb  = alloc_bf((size_t)LN * EE);
  bf16_t* oyb  = alloc_bf((size_t)LN * EE);

  auto cvt = [&](const float* s, bf16_t* d, int n) {
    cvt_bf16_kernel<<<(n + 255) / 256, 256, 0, stream>>>(s, d, n);
  };
  cvt(x, xb, LN * EE);
  cvt(y, yb, LN * EE);
  cvt(in_w_x, wixb, E3 * EE);
  cvt(in_w_y, wiyb, E3 * EE);
  cvt(out_w_x, woxb, EE * EE);
  cvt(out_w_y, woyb, EE * EE);

  // QKV projections (both streams) -> qcat/kcat (concat layout) + transposed V
  qkv_kernel<<<dim3(E3 / 64, LN / 32), 32, 0, stream>>>(xb, wixb, in_b_x, qcat, kcat, vtx, 0);
  qkv_kernel<<<dim3(E3 / 64, LN / 32), 32, 0, stream>>>(yb, wiyb, in_b_y, qcat, kcat, vty, DDIM);

  // flash attention with shared softmax across streams
  attn_kernel<<<dim3(LQ / 128, NHH), 128, 0, stream>>>(qcat, kcat, vtx, vty, oxb, oyb);

  // output projections
  outproj_kernel<<<dim3(EE / 64, LN / 32), 32, 0, stream>>>(oxb, woxb, out_b_x, out);
  outproj_kernel<<<dim3(EE / 64, LN / 32), 32, 0, stream>>>(oyb, woyb, out_b_y, out + (size_t)LN * EE);
}
